// GAT_72593537237728
// MI455X (gfx1250) — matrix operand order repaired and checked by reading, not compiled, not run on hardware
//
#include <hip/hip_runtime.h>

#define DIN   128
#define HC    128
#define NH    8
#define CH    16
#define DOUT  40
#define DPAD  48

typedef __attribute__((ext_vector_type(16))) __bf16        v16bf;
typedef __attribute__((ext_vector_type(8)))  float         v8f;
typedef __attribute__((ext_vector_type(4)))  unsigned int  u32x4;

union FragBF { v16bf v; u32x4 q[2]; };

// ---- helpers -------------------------------------------------------------

__device__ __forceinline__ unsigned short f2bf(float f) {
    unsigned u = __float_as_uint(f);
    u = u + 0x7FFFu + ((u >> 16) & 1u);   // round-to-nearest-even
    return (unsigned short)(u >> 16);
}

// order-preserving float -> uint map (for atomicMax on floats)
__device__ __forceinline__ unsigned fenc(float f) {
    unsigned u = __float_as_uint(f);
    return (u & 0x80000000u) ? ~u : (u | 0x80000000u);
}
__device__ __forceinline__ float fdec(unsigned u) {
    return __uint_as_float((u & 0x80000000u) ? (u ^ 0x80000000u) : ~u);
}

__device__ __forceinline__ void edge_sd(const int* __restrict__ ei, int E, int e,
                                        int& s, int& d) {
    if (e < E) { s = ei[e]; d = ei[E + e]; }
    else       { s = d = e - E; }            // self loops appended
}

// ---- conversion kernels --------------------------------------------------

__global__ void cvt_x_bf16(const float* __restrict__ in,
                           unsigned short* __restrict__ out, int n) {
    int t = blockIdx.x * blockDim.x + threadIdx.x;
    if (t < n) out[t] = f2bf(in[t]);
}

// W [K, Ncols] f32 row-major  ->  Wt [NcolsPad, K] bf16 (transposed, zero-padded)
__global__ void cvt_w_transpose(const float* __restrict__ W,
                                unsigned short* __restrict__ Wt,
                                int K, int Ncols, int NcolsPad) {
    int t = blockIdx.x * blockDim.x + threadIdx.x;
    if (t >= K * NcolsPad) return;
    int np = t / K, k = t % K;
    float v = (np < Ncols) ? W[(size_t)k * Ncols + np] : 0.0f;
    Wt[(size_t)np * K + k] = f2bf(v);
}

// ---- bf16 WMMA GEMM: C[N,Ncols] = A[N,128] * Bt[Ncols,128]^T -------------
// one wave -> 16x16 tile; blockDim = 32 * (Ncols/16); grid.x = N/16
__global__ void gemm_bf16_wmma(const unsigned short* __restrict__ A,
                               const unsigned short* __restrict__ Bt,
                               float* __restrict__ Cm, int Ncols) {
    const int lane = threadIdx.x & 31;
    const int wid  = threadIdx.x >> 5;
    const int row0 = blockIdx.x * 16;
    const int col0 = wid * 16;
    const int lm   = lane & 15;
    const int g    = lane >> 4;

    const unsigned short* __restrict__ arow = A  + (size_t)(row0 + lm) * DIN;
    const unsigned short* __restrict__ bcol = Bt + (size_t)(col0 + lm) * DIN;

    v8f acc = {};
#pragma unroll
    for (int kb = 0; kb < DIN; kb += 32) {
        FragBF fa, fb;
        fa.q[0] = *(const u32x4*)(arow + kb + 8 * g);
        fa.q[1] = *(const u32x4*)(arow + kb + 16 + 8 * g);
        fb.q[0] = *(const u32x4*)(bcol + kb + 8 * g);
        fb.q[1] = *(const u32x4*)(bcol + kb + 16 + 8 * g);
        acc = __builtin_amdgcn_wmma_f32_16x16x32_bf16(
            false, fa.v, false, fb.v, (short)0, acc, false, false);
    }

    float* __restrict__ crow = Cm + (size_t)row0 * Ncols + col0;
#pragma unroll
    for (int v = 0; v < 8; ++v) {
        int m = v + 8 * g;
        crow[(size_t)m * Ncols + lm] = acc[v];
    }
}

// ---- attention prep: al_s/al_d dot products + buffer init ----------------
__global__ void attn_prep(const float* __restrict__ xh,
                          const float* __restrict__ a_s,
                          const float* __restrict__ a_d,
                          float* __restrict__ als, float* __restrict__ ald,
                          unsigned* __restrict__ maxu, float* __restrict__ ssum,
                          float* __restrict__ accum, int N) {
    int t = blockIdx.x * blockDim.x + threadIdx.x;      // one per (n, h)
    if (t >= N * NH) return;
    int n = t >> 3, h = t & 7;
    const float* __restrict__ xr = xh + (size_t)n * HC + h * CH;
    const float* __restrict__ as = a_s + h * CH;
    const float* __restrict__ ad = a_d + h * CH;
    float s = 0.f, d = 0.f;
#pragma unroll
    for (int c = 0; c < CH; ++c) { s += xr[c] * as[c]; d += xr[c] * ad[c]; }
    als[t] = s; ald[t] = d;
    maxu[t] = 0u;          // decodes to most-negative
    ssum[t] = 0.f;
    float* __restrict__ ac = accum + (size_t)n * HC + h * CH;
#pragma unroll
    for (int c = 0; c < CH; ++c) ac[c] = 0.f;
}

// ---- edge pass 1: leaky-relu logits + segment max ------------------------
__global__ void edge_logits_max(const int* __restrict__ ei, int E, int Etot,
                                const float* __restrict__ als,
                                const float* __restrict__ ald,
                                float* __restrict__ ev,
                                unsigned* __restrict__ maxu) {
    int e = blockIdx.x * blockDim.x + threadIdx.x;
    if (e >= Etot) return;
    int s, d; edge_sd(ei, E, e, s, d);
#pragma unroll
    for (int h = 0; h < NH; ++h) {
        float v = als[s * NH + h] + ald[d * NH + h];
        v = v > 0.f ? v : 0.2f * v;
        ev[(size_t)e * NH + h] = v;
        atomicMax(&maxu[d * NH + h], fenc(v));
    }
}

// ---- edge pass 2: exp(logit - max) + segment sum -------------------------
__global__ void edge_exp_sum(const int* __restrict__ ei, int E, int Etot,
                             const unsigned* __restrict__ maxu,
                             float* __restrict__ ev,
                             float* __restrict__ ssum) {
    int e = blockIdx.x * blockDim.x + threadIdx.x;
    if (e >= Etot) return;
    int s, d; edge_sd(ei, E, e, s, d);
#pragma unroll
    for (int h = 0; h < NH; ++h) {
        float m = fdec(maxu[d * NH + h]);
        float x = __expf(ev[(size_t)e * NH + h] - m);
        ev[(size_t)e * NH + h] = x;
        atomicAdd(&ssum[d * NH + h], x);
    }
}

// ---- edge pass 3: alpha-weighted message scatter -------------------------
__global__ void edge_scatter(const int* __restrict__ ei, int E, int Etot,
                             const float* __restrict__ ev,
                             const float* __restrict__ ssum,
                             const float* __restrict__ xh,
                             float* __restrict__ accum) {
    int t = blockIdx.x * blockDim.x + threadIdx.x;      // one per (edge, h)
    if (t >= Etot * NH) return;
    int e = t >> 3, h = t & 7;
    int s, d; edge_sd(ei, E, e, s, d);
    float alpha = ev[t] / (ssum[d * NH + h] + 1e-16f);
    const float4* __restrict__ xs = (const float4*)(xh + (size_t)s * HC + h * CH);
    float* __restrict__ od = accum + (size_t)d * HC + h * CH;
#pragma unroll
    for (int q4 = 0; q4 < 4; ++q4) {
        float4 v = xs[q4];
        atomicAdd(&od[q4 * 4 + 0], alpha * v.x);
        atomicAdd(&od[q4 * 4 + 1], alpha * v.y);
        atomicAdd(&od[q4 * 4 + 2], alpha * v.z);
        atomicAdd(&od[q4 * 4 + 3], alpha * v.w);
    }
}

// ---- finalize: +bias, ELU, emit bf16 for next GEMM -----------------------
__global__ void finalize_elu(const float* __restrict__ accum,
                             const float* __restrict__ bias,
                             unsigned short* __restrict__ out_bf, int N) {
    int t = blockIdx.x * blockDim.x + threadIdx.x;
    if (t >= N * HC) return;
    float v = accum[t] + bias[t & (HC - 1)];
    v = v > 0.f ? v : (__expf(v) - 1.f);
    out_bf[t] = f2bf(v);
}

// ---- classifier log-softmax: one wave per row (40 of 48 padded cols) -----
__global__ void log_softmax40(const float* __restrict__ clog,
                              const float* __restrict__ bl,
                              float* __restrict__ out, int N) {
    int gt   = blockIdx.x * blockDim.x + threadIdx.x;
    int row  = gt >> 5;
    int lane = gt & 31;
    if (row >= N) return;
    const float* __restrict__ r = clog + (size_t)row * DPAD;
    float x0 = r[lane] + bl[lane];
    float x1 = (lane < DOUT - 32) ? (r[lane + 32] + bl[lane + 32]) : -3.4e38f;
    float mx = fmaxf(x0, x1);
#pragma unroll
    for (int o = 16; o > 0; o >>= 1) mx = fmaxf(mx, __shfl_xor(mx, o, 32));
    float s = __expf(x0 - mx) + ((lane < DOUT - 32) ? __expf(x1 - mx) : 0.f);
#pragma unroll
    for (int o = 16; o > 0; o >>= 1) s += __shfl_xor(s, o, 32);
    float ls = __logf(s);
    out[(size_t)row * DOUT + lane] = x0 - mx - ls;
    if (lane < DOUT - 32)
        out[(size_t)row * DOUT + lane + 32] = x1 - mx - ls;
}

// ---- host orchestration --------------------------------------------------

extern "C" void kernel_launch(void* const* d_in, const int* in_sizes, int n_in,
                              void* d_out, int out_size, void* d_ws, size_t ws_size,
                              hipStream_t stream) {
    const float* x   = (const float*)d_in[0];
    const int*   ei  = (const int*)  d_in[1];
    const float* W1  = (const float*)d_in[2];
    const float* as1 = (const float*)d_in[3];
    const float* ad1 = (const float*)d_in[4];
    const float* b1  = (const float*)d_in[5];
    const float* W2  = (const float*)d_in[6];
    const float* as2 = (const float*)d_in[7];
    const float* ad2 = (const float*)d_in[8];
    const float* b2  = (const float*)d_in[9];
    const float* Wl  = (const float*)d_in[10];
    const float* bl  = (const float*)d_in[11];

    const int N    = in_sizes[0] / DIN;
    const int E    = in_sizes[1] / 2;
    const int Etot = E + N;

    // carve workspace
    char* p = (char*)d_ws;
    auto carve = [&](size_t bytes) -> void* {
        void* r = (void*)p; p += (bytes + 255) & ~(size_t)255; return r;
    };
    unsigned short* x_bf  = (unsigned short*)carve((size_t)N * DIN * 2);
    unsigned short* h_bf  = (unsigned short*)carve((size_t)N * HC  * 2);
    unsigned short* Wt1   = (unsigned short*)carve((size_t)DIN * HC * 2);
    unsigned short* Wt2   = (unsigned short*)carve((size_t)HC  * HC * 2);
    unsigned short* Wtl   = (unsigned short*)carve((size_t)DPAD * HC * 2);
    float*    xh    = (float*)   carve((size_t)N * HC * 4);
    float*    als   = (float*)   carve((size_t)N * NH * 4);
    float*    ald   = (float*)   carve((size_t)N * NH * 4);
    unsigned* maxu  = (unsigned*)carve((size_t)N * NH * 4);
    float*    ssum  = (float*)   carve((size_t)N * NH * 4);
    float*    accum = (float*)   carve((size_t)N * HC * 4);
    float*    ev    = (float*)   carve((size_t)Etot * NH * 4);
    float*    clog  = (float*)   carve((size_t)N * DPAD * 4);

    const int BT = 256;
    auto cdiv = [](int a, int b) { return (a + b - 1) / b; };

    // convert inputs to bf16 (x) / transposed bf16 (weights)
    cvt_x_bf16<<<cdiv(N * DIN, BT), BT, 0, stream>>>(x, x_bf, N * DIN);
    cvt_w_transpose<<<cdiv(DIN * HC, BT), BT, 0, stream>>>(W1, Wt1, DIN, HC, HC);
    cvt_w_transpose<<<cdiv(HC * HC, BT), BT, 0, stream>>>(W2, Wt2, HC, HC, HC);
    cvt_w_transpose<<<cdiv(HC * DPAD, BT), BT, 0, stream>>>(Wl, Wtl, HC, DOUT, DPAD);

    auto gat_layer = [&](const unsigned short* in_bf, const unsigned short* Wt,
                         const float* a_s, const float* a_d, const float* bias,
                         unsigned short* out_bf) {
        gemm_bf16_wmma<<<N / 16, 32 * (HC / 16), 0, stream>>>(in_bf, Wt, xh, HC);
        attn_prep<<<cdiv(N * NH, BT), BT, 0, stream>>>(xh, a_s, a_d, als, ald,
                                                       maxu, ssum, accum, N);
        edge_logits_max<<<cdiv(Etot, BT), BT, 0, stream>>>(ei, E, Etot, als, ald, ev, maxu);
        edge_exp_sum<<<cdiv(Etot, BT), BT, 0, stream>>>(ei, E, Etot, maxu, ev, ssum);
        edge_scatter<<<cdiv(Etot * NH, BT), BT, 0, stream>>>(ei, E, Etot, ev, ssum, xh, accum);
        finalize_elu<<<cdiv(N * HC, BT), BT, 0, stream>>>(accum, bias, out_bf, N);
    };

    gat_layer(x_bf, Wt1, as1, ad1, b1, h_bf);   // layer 1
    gat_layer(h_bf, Wt2, as2, ad2, b2, h_bf);   // layer 2 (in-place on h_bf)

    // classifier GEMM (48 padded cols -> 3 waves/block) + log-softmax
    gemm_bf16_wmma<<<N / 16, 32 * (DPAD / 16), 0, stream>>>(h_bf, Wtl, clog, DPAD);
    log_softmax40<<<cdiv(N * 32, BT), BT, 0, stream>>>(clog, bl, (float*)d_out, N);
}
